// HierarchicalVQEncoder_80616536146015
// MI455X (gfx1250) — matrix operand order repaired and checked by reading, not compiled, not run on hardware
//
#include <hip/hip_runtime.h>
#include <hip/hip_bf16.h>

typedef __attribute__((ext_vector_type(16))) _Float16 v16h;
typedef __attribute__((ext_vector_type(8)))  float    v8f;
typedef __attribute__((ext_vector_type(4)))  unsigned u32x4;
typedef __attribute__((ext_vector_type(8)))  unsigned u32x8;

// ---------------- dimensions ----------------
#define BATCH 16
#define TLEN  2048
#define BT    (BATCH*TLEN)   // 32768
#define DIN   1024
#define DD    256
#define DG    128
#define DB    64
#define NL1   8
#define NL2   32

// ---------------- WMMA helpers (ISA 7.12.2 layouts) ----------------
__device__ inline v8f wmma16(v16h a, v16h b, v8f c) {
  // v_wmma_f32_16x16x32_f16
  return __builtin_amdgcn_wmma_f32_16x16x32_f16(false, a, false, b, (short)0, c, false, false);
}

// A: 16x32 (MxK) f16, row-major source with leading dim lda.
// lanes 0-15: row=lane, K = {0..7}+{16..23}; lanes 16-31: row=lane-16, K = {8..15}+{24..31}
__device__ inline v16h load_a_frag(const _Float16* A, int lda) {
  int l   = threadIdx.x & 31;
  int row = l & 15;
  int koff = (l >> 4) << 3;        // 0 or 8
  const _Float16* p = A + row * lda;
  v16h a;
#pragma unroll
  for (int e = 0; e < 8; ++e) {
    a[e]     = p[koff + e];
    a[e + 8] = p[16 + koff + e];
  }
  return a;
}

// Same fragment but from an f32 source (LDS tile DMA'd by the TDM), convert in VALU.
__device__ inline v16h load_a_frag_f32(const float* A, int lda) {
  int l   = threadIdx.x & 31;
  int row = l & 15;
  int koff = (l >> 4) << 3;
  const float* p = A + row * lda;
  v16h a;
#pragma unroll
  for (int e = 0; e < 8; ++e) {
    a[e]     = (_Float16)p[koff + e];
    a[e + 8] = (_Float16)p[16 + koff + e];
  }
  return a;
}

// B: 32x16 (KxN) where B[k][n] = W[n][k], W row-major (out,in) with leading dim ldw.
// lane l holds column n=l&15; lanes 0-15 K={0..7}+{16..23}, lanes 16-31 K={8..15}+{24..31}
__device__ inline v16h load_b_frag(const _Float16* W, int ldw) {
  int l  = threadIdx.x & 31;
  int n  = l & 15;
  int kb = (l >> 4) << 3;          // 0 or 8
  const _Float16* p = W + n * ldw + kb;
  v16h b;
#pragma unroll
  for (int e = 0; e < 8; ++e) { b[e] = p[e]; b[e + 8] = p[16 + e]; }
  return b;
}

// C/D: VGPR v -> m = v + (lane>=16 ? 8 : 0), n = lane&15
__device__ inline void store_c_lds(float* C, int ldc, v8f c, const float* bias, int nbias) {
  int l  = threadIdx.x & 31;
  int n  = l & 15;
  int mb = (l >> 4) << 3;
#pragma unroll
  for (int v = 0; v < 8; ++v) C[(mb + v) * ldc + n] = c[v] + bias[nbias + n];
}

// ---------------- Tensor Data Mover: 2D tile DMA global -> LDS ----------------
// Builds the D# per ISA 08_async_tensor §8 and issues TENSOR_LOAD_TO_LDS with
// two SGPR descriptor groups (2D tensor: VADDR2/3 omitted -> NULL).
// dsz_log2: 0=1B, 1=2B, 2=4B. Caller (one wave) must s_wait_tensorcnt then barrier.
__device__ inline void tdm_load_2d(const void* gsrc, void* ldst,
                                   unsigned rows, unsigned cols,
                                   unsigned stride_elems, unsigned dsz_log2) {
  unsigned long long ga = (unsigned long long)(size_t)gsrc;
  unsigned la = (unsigned)(size_t)ldst;     // generic LDS ptr low 32 bits = LDS offset
  unsigned td0 = stride_elems;              // tensor dim0 length (OOB bound)
  unsigned td1 = rows;                      // tensor dim1 length
  u32x4 g0;
  g0[0] = 1u;                                                     // count=1, user D#
  g0[1] = la;                                                     // lds_addr
  g0[2] = (unsigned)ga;                                           // global_addr[31:0]
  g0[3] = (unsigned)((ga >> 32) & 0x01FFFFFFull) | (2u << 30);    // addr[56:32] | type=2
  u32x8 g1;
  g1[0] = dsz_log2 << 16;                                         // mask=0, data_size
  g1[1] = (td0 & 0xFFFFu) << 16;                                  // tensor_dim0 lo
  g1[2] = (td0 >> 16) | ((td1 & 0xFFFFu) << 16);                  // dim0 hi | dim1 lo
  g1[3] = (td1 >> 16) | (cols << 16);                             // dim1 hi | tile_dim0
  g1[4] = rows;                                                   // tile_dim1 | tile_dim2=0
  g1[5] = stride_elems;                                           // tensor_dim0_stride lo
  g1[6] = 0u;                                                     // stride0 hi | stride1 lo
  g1[7] = 0u;                                                     // stride1 hi
  asm volatile("tensor_load_to_lds %0, %1" :: "s"(g0), "s"(g1) : "memory");
}

// ---------------- K0: g_in = mean_T(x) + std_T(x, ddof=1) ----------------
__global__ void k_colstats(const float* __restrict__ x, float* __restrict__ g_in) {
  int gid = blockIdx.x * 256 + threadIdx.x;       // 16*1024 pairs
  int b = gid >> 10, d = gid & 1023;
  const float* p = x + (size_t)b * TLEN * DIN + d;
  float s = 0.f, s2 = 0.f;
  for (int t = 0; t < TLEN; ++t) {
    float v = p[(size_t)t * DIN];
    s += v; s2 += v * v;
  }
  float mean = s * (1.f / TLEN);
  float var  = fmaxf((s2 - s * mean) * (1.f / (TLEN - 1)), 0.f);
  g_in[b * DIN + d] = mean + sqrtf(var);
}

// ---------------- K1: generic f32 -> f16 (strided rows) ----------------
__global__ void k_cvt(const float* __restrict__ src, _Float16* __restrict__ dst,
                      int rows, int src_ld, int cols) {
  int i = blockIdx.x * 256 + threadIdx.x;
  if (i < rows * cols) {
    int r = i / cols, c = i % cols;
    dst[i] = (_Float16)src[r * src_ld + c];
  }
}

// ---------------- K2: per-batch global path ----------------
__device__ inline float block_reduce_sum(float v, float* red) {
  int t = threadIdx.x;
  red[t] = v; __syncthreads();
  for (int s = 128; s > 0; s >>= 1) { if (t < s) red[t] += red[t + s]; __syncthreads(); }
  float r = red[0]; __syncthreads();
  return r;
}

__global__ void k_global(const float* __restrict__ g_in,
                         const float* __restrict__ Wg1, const float* __restrict__ bg1,
                         const float* __restrict__ Wg2, const float* __restrict__ bg2,
                         const float* __restrict__ Wp,  const float* __restrict__ bp,
                         const float* __restrict__ cb1,
                         const float* __restrict__ Wf,  const float* __restrict__ bf,
                         const float* __restrict__ cb2,
                         int* __restrict__ idx1_out, float* __restrict__ fconst,
                         float* __restrict__ cb2n) {
  __shared__ float red[256];
  __shared__ float h[256];
  __shared__ float gpv[128];
  __shared__ float h1[128];
  __shared__ float l1e[128];
  __shared__ float logits[NL1];
  __shared__ int   s_idx;
  int b = blockIdx.x, t = threadIdx.x;
  const float* gi = g_in + b * DIN;

  // h = relu(ln(g_in @ Wg1^T + bg1))
  {
    float acc = 0.f;
    const float* w = Wg1 + t * DIN;
    for (int k = 0; k < DIN; ++k) acc += gi[k] * w[k];
    h[t] = acc + bg1[t];
  }
  __syncthreads();
  {
    float sum  = block_reduce_sum(h[t], red);
    float mean = sum * (1.f / 256.f);
    float d    = h[t] - mean;
    float var  = block_reduce_sum(d * d, red) * (1.f / 256.f);
    float rstd = rsqrtf(var + 1e-5f);
    float hr   = d * rstd; hr = hr > 0.f ? hr : 0.f;
    __syncthreads();
    h[t] = hr;
  }
  __syncthreads();
  // gp = h @ Wg2^T + bg2  (128)
  if (t < DG) {
    float acc = 0.f;
    const float* w = Wg2 + t * DD;
    for (int k = 0; k < DD; ++k) acc += h[k] * w[k];
    gpv[t] = acc + bg2[t];
  }
  __syncthreads();
  // h1 = ln(gp @ Wp^T + bp)
  if (t < DG) {
    float acc = 0.f;
    const float* w = Wp + t * DG;
    for (int k = 0; k < DG; ++k) acc += gpv[k] * w[k];
    h1[t] = acc + bp[t];
  }
  __syncthreads();
  {
    float v1   = (t < DG) ? h1[t] : 0.f;
    float sum  = block_reduce_sum(v1, red);
    float mean = sum * (1.f / 128.f);
    float d    = (t < DG) ? (h1[t] - mean) : 0.f;
    float var  = block_reduce_sum(d * d, red) * (1.f / 128.f);
    float rstd = rsqrtf(var + 1e-5f);
    if (t < DG) h1[t] = d * rstd;
  }
  __syncthreads();
  // logits1 = l2norm(h1) . l2norm(cb1_k)
  if (t < NL1) {
    const float* c = cb1 + t * DG;
    float num = 0.f, cn = 0.f, hn = 0.f;
    for (int k = 0; k < DG; ++k) { num += h1[k] * c[k]; cn += c[k] * c[k]; hn += h1[k] * h1[k]; }
    logits[t] = num / (fmaxf(sqrtf(hn), 1e-12f) * fmaxf(sqrtf(cn), 1e-12f));
  }
  __syncthreads();
  if (t == 0) {
    int bi = 0; float bv = logits[0];
    for (int k = 1; k < NL1; ++k) if (logits[k] > bv) { bv = logits[k]; bi = k; }
    s_idx = bi; idx1_out[b] = bi;
  }
  __syncthreads();
  int idx = s_idx;
  if (t < DG) l1e[t] = cb1[idx * DG + t];
  __syncthreads();
  // fconst = Wf[:,256:384] @ l1_emb + bf
  {
    float acc = 0.f;
    const float* w = Wf + t * (DD + DG) + DD;
    for (int k = 0; k < DG; ++k) acc += l1e[k] * w[k];
    fconst[b * DD + t] = acc + bf[t];
  }
  // normalized per-batch L2 codebook
  if (t < NL2) {
    const float* c = cb2 + ((size_t)idx * NL2 + t) * DB;
    float n2 = 0.f;
    for (int k = 0; k < DB; ++k) n2 += c[k] * c[k];
    float inv = 1.f / fmaxf(sqrtf(n2), 1e-12f);
    float* o = cb2n + (b * NL2 + t) * DB;
    for (int k = 0; k < DB; ++k) o[k] = c[k] * inv;
  }
}

// ---------------- K3: a1 = f16(relu(ln(x @ Wl1^T + bl1))) ----------------
// Whole 16x1024 f32 A-tile DMA'd to LDS by the TDM once; K-loop is barrier-free.
__global__ void k_gemm1(const float* __restrict__ x, const _Float16* __restrict__ Wl1h,
                        const float* __restrict__ bl1, _Float16* __restrict__ a1) {
  __shared__ float As[16 * DIN];   // 64 KB
  __shared__ float Cs[16 * DD];    // 16 KB
  __shared__ float mstat[16], rstat[16];
  int row0 = blockIdx.x * 16;
  int wave = threadIdx.x >> 5;
  if (threadIdx.x < 32) {          // one wave issues the DMA (TDM ignores EXEC)
    tdm_load_2d(x + (size_t)row0 * DIN, As, 16, DIN, DIN, 2u /*4B*/);
    __builtin_amdgcn_s_wait_tensorcnt(0);
  }
  __syncthreads();
  v8f c0 = {}, c1 = {};
  for (int kk = 0; kk < DIN; kk += 32) {
    v16h a  = load_a_frag_f32(As + kk, DIN);
    v16h b0 = load_b_frag(Wl1h + (size_t)(wave * 32)      * DIN + kk, DIN);
    v16h b1 = load_b_frag(Wl1h + (size_t)(wave * 32 + 16) * DIN + kk, DIN);
    c0 = wmma16(a, b0, c0);
    c1 = wmma16(a, b1, c1);
  }
  store_c_lds(Cs + wave * 32,      DD, c0, bl1, wave * 32);
  store_c_lds(Cs + wave * 32 + 16, DD, c1, bl1, wave * 32 + 16);
  __syncthreads();
  if (threadIdx.x < 16) {
    const float* r = Cs + threadIdx.x * DD;
    float s = 0.f; for (int k = 0; k < DD; ++k) s += r[k];
    float m = s * (1.f / DD);
    float v = 0.f; for (int k = 0; k < DD; ++k) { float d = r[k] - m; v += d * d; }
    mstat[threadIdx.x] = m;
    rstat[threadIdx.x] = rsqrtf(v * (1.f / DD) + 1e-5f);
  }
  __syncthreads();
  for (int i = threadIdx.x; i < 16 * DD; i += 256) {
    int r = i >> 8, n = i & 255;
    float v = (Cs[i] - mstat[r]) * rstat[r];
    v = v > 0.f ? v : 0.f;
    a1[(size_t)(row0 + r) * DD + n] = (_Float16)v;
  }
}

// ---------------- K4: lp = a1 @ Wl2^T + bl2 (f16 out) ----------------
__global__ void k_gemm2(const _Float16* __restrict__ a1, const _Float16* __restrict__ Wl2h,
                        const float* __restrict__ bl2, _Float16* __restrict__ lph) {
  __shared__ _Float16 At[16 * DD];   // 8 KB via TDM
  int row0 = blockIdx.x * 16;
  int wave = threadIdx.x >> 5;
  if (threadIdx.x < 32) {
    tdm_load_2d(a1 + (size_t)row0 * DD, At, 16, DD, DD, 1u /*2B*/);
    __builtin_amdgcn_s_wait_tensorcnt(0);
  }
  __syncthreads();
  v8f c0 = {}, c1 = {};
  for (int kk = 0; kk < DD; kk += 32) {
    v16h a  = load_a_frag(At + kk, DD);
    v16h b0 = load_b_frag(Wl2h + (size_t)(wave * 32)      * DD + kk, DD);
    v16h b1 = load_b_frag(Wl2h + (size_t)(wave * 32 + 16) * DD + kk, DD);
    c0 = wmma16(a, b0, c0);
    c1 = wmma16(a, b1, c1);
  }
  int l = threadIdx.x & 31, n = l & 15, mb = (l >> 4) << 3;
  int n0 = wave * 32;
#pragma unroll
  for (int v = 0; v < 8; ++v) {
    int m = mb + v;
    lph[(size_t)(row0 + m) * DD + n0 + n]      = (_Float16)(c0[v] + bl2[n0 + n]);
    lph[(size_t)(row0 + m) * DD + n0 + 16 + n] = (_Float16)(c1[v] + bl2[n0 + 16 + n]);
  }
}

// ---------------- K5: hb = ln(lp @ Wtb^T + btb); L2-VQ argmax; emb_low ----------------
__global__ void k_hb_vq(const _Float16* __restrict__ lph, const _Float16* __restrict__ Wtbh,
                        const float* __restrict__ btb, const float* __restrict__ cb2n,
                        const float* __restrict__ cb2, const int* __restrict__ idx1,
                        _Float16* __restrict__ embl) {
  __shared__ _Float16 At[32 * DD];   // 16 KB via TDM
  __shared__ float Hs[32 * DB];
  __shared__ float best[32 * 8];
  __shared__ int   bidx[32 * 8];
  __shared__ int   rowcode[32];
  int row0 = blockIdx.x * 32;
  int bat  = row0 >> 11;                  // 2048 rows per batch
  int wave = threadIdx.x >> 5;
  int mt = wave >> 2, nt = wave & 3;      // 2 M-tiles x 4 N-tiles
  if (threadIdx.x < 32) {
    tdm_load_2d(lph + (size_t)row0 * DD, At, 32, DD, DD, 1u /*2B*/);
    __builtin_amdgcn_s_wait_tensorcnt(0);
  }
  __syncthreads();
  v8f c = {};
  for (int kk = 0; kk < DD; kk += 32) {
    v16h a   = load_a_frag(At + (size_t)(mt * 16) * DD + kk, DD);
    v16h bf_ = load_b_frag(Wtbh + (size_t)(nt * 16) * DD + kk, DD);
    c = wmma16(a, bf_, c);
  }
  { int l = threadIdx.x & 31, n = l & 15, mb = (l >> 4) << 3;
#pragma unroll
    for (int v = 0; v < 8; ++v)
      Hs[(mt * 16 + mb + v) * DB + nt * 16 + n] = c[v] + btb[nt * 16 + n]; }
  __syncthreads();
  if (threadIdx.x < 32) {     // LN per row over 64
    float* r = Hs + threadIdx.x * DB;
    float s = 0.f; for (int k = 0; k < DB; ++k) s += r[k];
    float m = s * (1.f / DB);
    float v = 0.f; for (int k = 0; k < DB; ++k) { float d = r[k] - m; v += d * d; }
    float rs = rsqrtf(v * (1.f / DB) + 1e-5f);
    for (int k = 0; k < DB; ++k) r[k] = (r[k] - m) * rs;
  }
  __syncthreads();
  { // argmax over 32 codes (positive scaling of hb doesn't change argmax)
    int r = threadIdx.x >> 3, g = threadIdx.x & 7;
    const float* hr = Hs + r * DB;
    float bm = -1e30f; int bi = 0;
    for (int c4 = 0; c4 < 4; ++c4) {
      int k = g * 4 + c4;
      const float* cn = cb2n + (bat * NL2 + k) * DB;
      float dt = 0.f;
      for (int j = 0; j < DB; ++j) dt += hr[j] * cn[j];
      if (dt > bm) { bm = dt; bi = k; }
    }
    best[r * 8 + g] = bm; bidx[r * 8 + g] = bi;
  }
  __syncthreads();
  if (threadIdx.x < 32) {
    int r = threadIdx.x;
    float bm = best[r * 8]; int bi = bidx[r * 8];
    for (int g = 1; g < 8; ++g) if (best[r * 8 + g] > bm) { bm = best[r * 8 + g]; bi = bidx[r * 8 + g]; }
    rowcode[r] = bi;
  }
  __syncthreads();
  int cb_base = idx1[bat] * NL2;
  for (int i = threadIdx.x; i < 32 * DB; i += 256) {
    int r = i >> 6, j = i & 63;
    embl[(size_t)(row0 + r) * DB + j] = (_Float16)cb2[((size_t)cb_base + rowcode[r]) * DB + j];
  }
}

// ---------------- K6: out = relu(ln( ln(emb@Wfb^T+bfb) @ WfA^T + fconst )) ----------------
__global__ void k_final(const _Float16* __restrict__ embl, const _Float16* __restrict__ Wfbh,
                        const float* __restrict__ bfb, const _Float16* __restrict__ WfAh,
                        const float* __restrict__ fconst, float* __restrict__ out) {
  __shared__ _Float16 Et[16 * DB];   // 2 KB via TDM
  __shared__ float    Cs[16 * DD];
  __shared__ _Float16 As2[16 * DD];
  __shared__ float    mstat[16], rstat[16];
  int row0 = blockIdx.x * 16;
  int bat  = row0 >> 11;
  int wave = threadIdx.x >> 5;
  if (threadIdx.x < 32) {
    tdm_load_2d(embl + (size_t)row0 * DB, Et, 16, DB, DB, 1u /*2B*/);
    __builtin_amdgcn_s_wait_tensorcnt(0);
  }
  __syncthreads();
  // stage 1: K = 64
  v8f c0 = {}, c1 = {};
  for (int kk = 0; kk < DB; kk += 32) {
    v16h a  = load_a_frag(Et + kk, DB);
    v16h b0 = load_b_frag(Wfbh + (size_t)(wave * 32)      * DB + kk, DB);
    v16h b1 = load_b_frag(Wfbh + (size_t)(wave * 32 + 16) * DB + kk, DB);
    c0 = wmma16(a, b0, c0);
    c1 = wmma16(a, b1, c1);
  }
  store_c_lds(Cs + wave * 32,      DD, c0, bfb, wave * 32);
  store_c_lds(Cs + wave * 32 + 16, DD, c1, bfb, wave * 32 + 16);
  __syncthreads();
  if (threadIdx.x < 16) {
    const float* r = Cs + threadIdx.x * DD;
    float s = 0.f; for (int k = 0; k < DD; ++k) s += r[k];
    float m = s * (1.f / DD);
    float v = 0.f; for (int k = 0; k < DD; ++k) { float d = r[k] - m; v += d * d; }
    mstat[threadIdx.x] = m;
    rstat[threadIdx.x] = rsqrtf(v * (1.f / DD) + 1e-5f);
  }
  __syncthreads();
  for (int i = threadIdx.x; i < 16 * DD; i += 256) {
    int r = i >> 8;
    As2[i] = (_Float16)((Cs[i] - mstat[r]) * rstat[r]);
  }
  __syncthreads();
  // stage 2: K = 256
  v8f d0 = {}, d1 = {};
  for (int kk = 0; kk < DD; kk += 32) {
    v16h a  = load_a_frag(As2 + kk, DD);
    v16h b0 = load_b_frag(WfAh + (size_t)(wave * 32)      * DD + kk, DD);
    v16h b1 = load_b_frag(WfAh + (size_t)(wave * 32 + 16) * DD + kk, DD);
    d0 = wmma16(a, b0, d0);
    d1 = wmma16(a, b1, d1);
  }
  __syncthreads();
  store_c_lds(Cs + wave * 32,      DD, d0, fconst + bat * DD, wave * 32);
  store_c_lds(Cs + wave * 32 + 16, DD, d1, fconst + bat * DD, wave * 32 + 16);
  __syncthreads();
  if (threadIdx.x < 16) {
    const float* r = Cs + threadIdx.x * DD;
    float s = 0.f; for (int k = 0; k < DD; ++k) s += r[k];
    float m = s * (1.f / DD);
    float v = 0.f; for (int k = 0; k < DD; ++k) { float d = r[k] - m; v += d * d; }
    mstat[threadIdx.x] = m;
    rstat[threadIdx.x] = rsqrtf(v * (1.f / DD) + 1e-5f);
  }
  __syncthreads();
  for (int i = threadIdx.x; i < 16 * DD; i += 256) {
    int r = i >> 8, n = i & 255;
    float v = (Cs[i] - mstat[r]) * rstat[r];
    v = v > 0.f ? v : 0.f;
    out[(size_t)(row0 + r) * DD + n] = v;
  }
}

// ---------------- host ----------------
extern "C" void kernel_launch(void* const* d_in, const int* in_sizes, int n_in,
                              void* d_out, int out_size, void* d_ws, size_t ws_size,
                              hipStream_t stream) {
  (void)in_sizes; (void)n_in; (void)out_size; (void)ws_size;
  const float* x   = (const float*)d_in[0];
  const float* Wg1 = (const float*)d_in[1];
  const float* bg1 = (const float*)d_in[2];
  const float* Wg2 = (const float*)d_in[3];
  const float* bg2 = (const float*)d_in[4];
  const float* Wl1 = (const float*)d_in[5];
  const float* bl1 = (const float*)d_in[6];
  const float* Wl2 = (const float*)d_in[7];
  const float* bl2 = (const float*)d_in[8];
  const float* Wp  = (const float*)d_in[9];
  const float* bp  = (const float*)d_in[10];
  const float* cb1 = (const float*)d_in[11];
  const float* Wtb = (const float*)d_in[12];
  const float* btb = (const float*)d_in[13];
  const float* Wfb = (const float*)d_in[14];
  const float* bfb = (const float*)d_in[15];
  const float* cb2 = (const float*)d_in[16];
  const float* Wf  = (const float*)d_in[17];
  const float* bf  = (const float*)d_in[18];
  float* out = (float*)d_out;

  char* ws = (char*)d_ws;
  size_t off = 0;
  auto take = [&](size_t bytes) -> char* {
    char* p = ws + off;
    off = (off + bytes + 255) & ~(size_t)255;
    return p;
  };
  float*    g_in   = (float*)take(BATCH * DIN * 4);
  int*      idx1   = (int*)take(BATCH * 4);
  float*    fconst = (float*)take(BATCH * DD * 4);
  float*    cb2n   = (float*)take(BATCH * NL2 * DB * 4);
  _Float16* Wl1h   = (_Float16*)take((size_t)DD * DIN * 2);
  _Float16* Wl2h   = (_Float16*)take((size_t)DD * DD * 2);
  _Float16* Wtbh   = (_Float16*)take((size_t)DB * DD * 2);
  _Float16* Wfbh   = (_Float16*)take((size_t)DD * DB * 2);
  _Float16* WfAh   = (_Float16*)take((size_t)DD * DD * 2);
  _Float16* a1     = (_Float16*)take((size_t)BT * DD * 2);
  _Float16* lph    = (_Float16*)take((size_t)BT * DD * 2);
  _Float16* embl   = (_Float16*)take((size_t)BT * DB * 2);

  k_colstats<<<(BATCH * DIN) / 256, 256, 0, stream>>>(x, g_in);
  k_cvt<<<(DD * DIN + 255) / 256, 256, 0, stream>>>(Wl1, Wl1h, DD, DIN, DIN);
  k_cvt<<<(DD * DD + 255) / 256, 256, 0, stream>>>(Wl2, Wl2h, DD, DD, DD);
  k_cvt<<<(DB * DD + 255) / 256, 256, 0, stream>>>(Wtb, Wtbh, DB, DD, DD);
  k_cvt<<<(DD * DB + 255) / 256, 256, 0, stream>>>(Wfb, Wfbh, DD, DB, DB);
  k_cvt<<<(DD * DD + 255) / 256, 256, 0, stream>>>(Wf, WfAh, DD, DD + DG, DD);
  k_global<<<BATCH, 256, 0, stream>>>(g_in, Wg1, bg1, Wg2, bg2, Wp, bp, cb1,
                                      Wf, bf, cb2, idx1, fconst, cb2n);
  k_gemm1<<<BT / 16, 256, 0, stream>>>(x, Wl1h, bl1, a1);
  k_gemm2<<<BT / 16, 256, 0, stream>>>(a1, Wl2h, bl2, lph);
  k_hb_vq<<<BT / 32, 256, 0, stream>>>(lph, Wtbh, btb, cb2n, cb2, idx1, embl);
  k_final<<<BT / 16, 256, 0, stream>>>(embl, Wfbh, bfb, WfAh, fconst, out);
}
